// Attn_52776558133289
// MI455X (gfx1250) — compile-verified, matrix operand order checked by reading, not run on hardware
//
#include <hip/hip_runtime.h>
#include <hip/hip_bf16.h>
#include <math.h>
#include <stdint.h>

// Problem constants (match the reference).
#define BB 32
#define SS 2048
#define EE 1024
#define HH 1024

typedef __attribute__((ext_vector_type(16))) __bf16 v16bf;
typedef __attribute__((ext_vector_type(8)))  __bf16 v8bf;
typedef __attribute__((ext_vector_type(8)))  float  v8f;

static __device__ __forceinline__ v16bf bf_combine(v8bf lo, v8bf hi) {
    v16bf r;
#pragma unroll
    for (int i = 0; i < 8; ++i) { r[i] = lo[i]; r[i + 8] = hi[i]; }
    return r;
}

// Native v_tanh_f32 (confirmed emitted on gfx1250); branch-free fallback.
static __device__ __forceinline__ float fast_tanh(float x) {
#if __has_builtin(__builtin_amdgcn_tanhf)
    return __builtin_amdgcn_tanhf(x);
#else
    float t = __expf(x + x);
    return 1.0f - 2.0f / (t + 1.0f);
#endif
}

// ---------------------------------------------------------------------------
// K1a: convert M_w[:, :E] (H x (E+H) f32, row-major) -> bf16 W [H x E]
// ---------------------------------------------------------------------------
__global__ __launch_bounds__(256) void attn_prep_w(const float* __restrict__ Mw,
                                                   __bf16* __restrict__ Wbf) {
    int idx = blockIdx.x * 256 + threadIdx.x;          // H*E = 1M threads
    int h = idx >> 10, e = idx & 1023;
    Wbf[idx] = (__bf16)Mw[(size_t)h * (EE + HH) + e];
}

// ---------------------------------------------------------------------------
// K1b: hproj[b][h] = M_b[h] + sum_k hidden[b][k] * M_w[h][E+k]
// ---------------------------------------------------------------------------
__global__ __launch_bounds__(256) void attn_prep_hproj(const float* __restrict__ hidden,
                                                       const float* __restrict__ Mw,
                                                       const float* __restrict__ Mb,
                                                       float* __restrict__ hproj) {
    int idx = blockIdx.x * 256 + threadIdx.x;          // B*H = 32768 threads
    int b = idx >> 10, h = idx & 1023;
    const float* hrow = hidden + (size_t)b * HH;
    const float* wrow = Mw + (size_t)h * (EE + HH) + EE;
    float acc = Mb[h];
#pragma unroll 8
    for (int k = 0; k < HH; ++k) acc = fmaf(hrow[k], wrow[k], acc);
    hproj[idx] = acc;
}

// ---------------------------------------------------------------------------
// K2: fused score GEMM.  Per workgroup: one (b, 32-row s-tile).
//     Each wave covers 128 H-columns, processing TWO 16-col n-tiles per
//     inner loop so every A-fragment LDS read feeds two WMMAs.
// ---------------------------------------------------------------------------
__global__ __launch_bounds__(256) void attn_score_wmma(const float* __restrict__ enc,
                                                       const __bf16* __restrict__ Wbf,
                                                       const float* __restrict__ hproj,
                                                       const float* __restrict__ Vw,
                                                       float* __restrict__ scores_p) {
    __shared__ __bf16 As[32 * 1024];                   // 64 KB, bf16, XOR-swizzled

    const int b  = blockIdx.y;
    const int s0 = blockIdx.x * 32;
    const int t  = threadIdx.x;

    // ---- stage 32 x 1024 enc tile, f32 -> bf16, swizzle 8-half groups ----
    const float* etile = enc + ((size_t)b * SS + s0) * EE;
    for (int g = t; g < 32 * 128; g += 256) {
        int row = g >> 7, c = g & 127;
        const float4* src = (const float4*)(etile + (size_t)row * EE + c * 8);
        float4 f0 = src[0], f1 = src[1];
        v8bf o;
        o[0] = (__bf16)f0.x; o[1] = (__bf16)f0.y; o[2] = (__bf16)f0.z; o[3] = (__bf16)f0.w;
        o[4] = (__bf16)f1.x; o[5] = (__bf16)f1.y; o[6] = (__bf16)f1.z; o[7] = (__bf16)f1.w;
        *(v8bf*)&As[row * 1024 + ((c ^ (row & 15)) << 3)] = o;
    }
    __syncthreads();

    const int wave = t >> 5;
    const int lane = t & 31;
    const int nsel = lane & 15;                        // N column / A row in tile
    const int hi   = lane >> 4;                        // K-half select
    const int hbase = wave * 128;                      // 8 waves x 128 H cols

    float sc[2][8];
#pragma unroll
    for (int mt = 0; mt < 2; ++mt)
#pragma unroll
        for (int r = 0; r < 8; ++r) sc[mt][r] = 0.0f;

    const int row0 = nsel, row1 = 16 + nsel;
    const int sw0 = row0 & 15, sw1 = row1 & 15;

    for (int np = 0; np < 4; ++np) {                   // pairs of n-tiles
        const int h0 = hbase + np * 32;
        const float hpA = hproj[(size_t)b * HH + h0 + nsel];
        const float hpB = hproj[(size_t)b * HH + h0 + 16 + nsel];
        v8f acc00, acc01, acc10, acc11;
#pragma unroll
        for (int r = 0; r < 8; ++r) {
            acc00[r] = hpA; acc01[r] = hpA;
            acc10[r] = hpB; acc11[r] = hpB;
        }

        const __bf16* wrowA = Wbf + (size_t)(h0 + nsel) * EE;
        const __bf16* wrowB = wrowA + (size_t)16 * EE;

        for (int ks = 0; ks < EE; ks += 32) {
            // B fragments (32x16 bf16) for the two n-tiles
            v16bf bA = bf_combine(*(const v8bf*)(wrowA + ks + hi * 8),
                                  *(const v8bf*)(wrowA + ks + hi * 8 + 16));
            v16bf bB = bf_combine(*(const v8bf*)(wrowB + ks + hi * 8),
                                  *(const v8bf*)(wrowB + ks + hi * 8 + 16));

            // A fragments (16x32 bf16) for the two m-tiles, from swizzled LDS
            int g0 = (ks >> 3) + hi;
            v16bf a0 = bf_combine(*(v8bf*)&As[row0 * 1024 + (((g0)     ^ sw0) << 3)],
                                  *(v8bf*)&As[row0 * 1024 + (((g0 + 2) ^ sw0) << 3)]);
            v16bf a1 = bf_combine(*(v8bf*)&As[row1 * 1024 + (((g0)     ^ sw1) << 3)],
                                  *(v8bf*)&As[row1 * 1024 + (((g0 + 2) ^ sw1) << 3)]);

            acc00 = __builtin_amdgcn_wmma_f32_16x16x32_bf16(
                false, a0, false, bA, (short)0, acc00, false, false);
            acc01 = __builtin_amdgcn_wmma_f32_16x16x32_bf16(
                false, a1, false, bA, (short)0, acc01, false, false);
            acc10 = __builtin_amdgcn_wmma_f32_16x16x32_bf16(
                false, a0, false, bB, (short)0, acc10, false, false);
            acc11 = __builtin_amdgcn_wmma_f32_16x16x32_bf16(
                false, a1, false, bB, (short)0, acc11, false, false);
        }

        const float vwA = Vw[h0 + nsel];
        const float vwB = Vw[h0 + 16 + nsel];
#pragma unroll
        for (int r = 0; r < 8; ++r) {
            sc[0][r] = fmaf(fast_tanh(acc00[r]), vwA, sc[0][r]);
            sc[1][r] = fmaf(fast_tanh(acc01[r]), vwA, sc[1][r]);
            sc[0][r] = fmaf(fast_tanh(acc10[r]), vwB, sc[0][r]);
            sc[1][r] = fmaf(fast_tanh(acc11[r]), vwB, sc[1][r]);
        }
    }

    // Butterfly-reduce across the 16 lanes of each half; lane group leaders store.
#pragma unroll
    for (int mt = 0; mt < 2; ++mt)
#pragma unroll
        for (int r = 0; r < 8; ++r) {
            float v = sc[mt][r];
            v += __shfl_xor(v, 1, 32);
            v += __shfl_xor(v, 2, 32);
            v += __shfl_xor(v, 4, 32);
            v += __shfl_xor(v, 8, 32);
            if (nsel == 0) {
                int srow = s0 + mt * 16 + (hi << 3) + r;   // lane16 holds rows m+8
                scores_p[((size_t)b * SS + srow) * 8 + wave] = v;
            }
        }
}

// ---------------------------------------------------------------------------
// K3: masked softmax over S per batch row.  weights -> d_out[B*E ...]
// ---------------------------------------------------------------------------
__global__ __launch_bounds__(256) void attn_softmax(const float* __restrict__ scores_p,
                                                    const unsigned char* __restrict__ mask,
                                                    const float* __restrict__ Vb,
                                                    float* __restrict__ weights) {
    __shared__ float svals[SS];
    __shared__ float red[256];
    const int b = blockIdx.x, t = threadIdx.x;
    const float vb = Vb[0];

    float lmax = -3.0e38f;
    for (int s = t; s < SS; s += 256) {
        const float4* p = (const float4*)(scores_p + ((size_t)b * SS + s) * 8);
        float4 p0 = p[0], p1 = p[1];
        float v = vb + ((p0.x + p0.y) + (p0.z + p0.w)) + ((p1.x + p1.y) + (p1.z + p1.w));
        if (mask[(size_t)b * SS + s]) v = -1.0e30f;    // reference: where(mask, NEG, x)
        svals[s] = v;
        lmax = fmaxf(lmax, v);
    }
    red[t] = lmax; __syncthreads();
    for (int o = 128; o > 0; o >>= 1) { if (t < o) red[t] = fmaxf(red[t], red[t + o]); __syncthreads(); }
    const float mx = red[0]; __syncthreads();

    float lsum = 0.0f;
    for (int s = t; s < SS; s += 256) {
        float e = __expf(svals[s] - mx);
        svals[s] = e;
        lsum += e;
    }
    red[t] = lsum; __syncthreads();
    for (int o = 128; o > 0; o >>= 1) { if (t < o) red[t] += red[t + o]; __syncthreads(); }
    const float inv = 1.0f / red[0];

    for (int s = t; s < SS; s += 256)
        weights[(size_t)b * SS + s] = svals[s] * inv;
}

// ---------------------------------------------------------------------------
// K4: weighted[b][e] = sum_s weights[b][s] * enc[b][s][e]   (memory bound)
// ---------------------------------------------------------------------------
__global__ __launch_bounds__(256) void attn_weighted(const float* __restrict__ enc,
                                                     const float* __restrict__ weights,
                                                     float* __restrict__ out) {
    const int b = blockIdx.x, t = threadIdx.x;
    const float* wrow = weights + (size_t)b * SS;
    const float* base = enc + (size_t)b * SS * EE + t * 4;
    float4 acc = make_float4(0.f, 0.f, 0.f, 0.f);
    for (int s = 0; s < SS; ++s) {
        __builtin_prefetch(base + (size_t)(s + 8) * EE, 0, 0);  // global_prefetch_b8
        float ws = wrow[s];
        float4 v = *(const float4*)(base + (size_t)s * EE);
        acc.x = fmaf(ws, v.x, acc.x);
        acc.y = fmaf(ws, v.y, acc.y);
        acc.z = fmaf(ws, v.z, acc.z);
        acc.w = fmaf(ws, v.w, acc.w);
    }
    *(float4*)(out + (size_t)b * EE + t * 4) = acc;
}

// ---------------------------------------------------------------------------
extern "C" void kernel_launch(void* const* d_in, const int* in_sizes, int n_in,
                              void* d_out, int out_size, void* d_ws, size_t ws_size,
                              hipStream_t stream) {
    const float*         hidden = (const float*)d_in[0];
    const float*         enc    = (const float*)d_in[1];
    const unsigned char* mask   = (const unsigned char*)d_in[2];  // jax bool -> 1 byte
    const float*         Mw     = (const float*)d_in[3];
    const float*         Mb     = (const float*)d_in[4];
    const float*         Vw     = (const float*)d_in[5];
    const float*         Vb     = (const float*)d_in[6];

    float* out      = (float*)d_out;
    float* weighted = out;                 // B*E floats
    float* weights  = out + BB * EE;       // B*S floats

    // Workspace layout
    char*   ws      = (char*)d_ws;
    __bf16* Wbf     = (__bf16*)ws;                               // H*E*2      = 2 MB
    float*  hproj   = (float*)(ws + (size_t)HH * EE * 2);        // B*H*4      = 128 KB
    float*  scoresp = (float*)(ws + (size_t)HH * EE * 2
                                   + (size_t)BB * HH * 4);       // B*S*8*4    = 2 MB

    attn_prep_w    <<<dim3((HH * EE) / 256), dim3(256), 0, stream>>>(Mw, Wbf);
    attn_prep_hproj<<<dim3((BB * HH) / 256), dim3(256), 0, stream>>>(hidden, Mw, Mb, hproj);
    attn_score_wmma<<<dim3(SS / 32, BB),     dim3(256), 0, stream>>>(enc, Wbf, hproj, Vw, scoresp);
    attn_softmax   <<<dim3(BB),              dim3(256), 0, stream>>>(scoresp, mask, Vb, weights);
    attn_weighted  <<<dim3(BB),              dim3(256), 0, stream>>>(enc, weights, weighted);
}
